// QuanConv2D_19164144075327
// MI455X (gfx1250) — compile-verified, matrix operand order checked by reading, not run on hardware
//
#include <hip/hip_runtime.h>

typedef __attribute__((ext_vector_type(16))) _Float16 v16h;
typedef __attribute__((ext_vector_type(8)))  _Float16 v8h;
typedef __attribute__((ext_vector_type(8)))  float    v8f;

#define NQ      9
#define DIM     512
#define MROWS   32                  // patches (rows) per workgroup
#define PSI_LD  (DIM + 8)           // halves; pad to spread LDS banks
#define PROB_LD (DIM + 4)           // floats

__device__ __forceinline__ v16h make_frag(v8h lo, v8h hi) {
    return __builtin_shufflevector(lo, hi,
           0, 1, 2, 3, 4, 5, 6, 7, 8, 9, 10, 11, 12, 13, 14, 15);
}

// ---------------------------------------------------------------------------
// Kernel 1: build the 512x512 complex unitary from the 27 weights, f16
// real/imag planes, laid out [n][k] row-major (the exact B-fragment access
// pattern for C = psi * U^T with v_wmma_f32_16x16x32_f16).
// ---------------------------------------------------------------------------
__global__ __launch_bounds__(256)
void build_u_kernel(const float* __restrict__ weight,
                    _Float16* __restrict__ Ur,
                    _Float16* __restrict__ Ui)
{
    __shared__ float g[NQ][2][2][2];   // [qubit][row][col][re/im]
    const int tid = threadIdx.x;

    if (tid < NQ) {
        const int q = tid;
        const float HM = 0.5f * 0.63245553203367586f;   // 0.5 * sqrt(2/5)
        float w0 = weight[3 * q + 0] * HM;
        float w1 = weight[3 * q + 1] * HM;
        float w2 = weight[3 * q + 2] * HM;
        float c0 = __cosf(w0), s0 = __sinf(w0);
        float c2 = __cosf(w2), s2 = __sinf(w2);
        float er = __cosf(w1), ei = -__sinf(w1);        // e = exp(-i*w1)
        float A = c2 * c0, B = s2 * s0, C = c2 * s0, D = s2 * c0;
        g[q][0][0][0] = (A - B) * er;  g[q][0][0][1] =  (A + B) * ei;
        g[q][0][1][0] = (C - D) * ei;  g[q][0][1][1] = -(C + D) * er;
        g[q][1][0][0] = (D - C) * ei;  g[q][1][0][1] = -(D + C) * er;
        g[q][1][1][0] = (A - B) * er;  g[q][1][1][1] = -(A + B) * ei;
    }
    __syncthreads();

    const int n = blockIdx.x;
    int r = n;
    #pragma unroll
    for (int q = NQ - 1; q >= 0; --q) {       // CNOT-ring row permutation
        int t = (q + 1) % NQ;
        r ^= ((r >> (NQ - 1 - q)) & 1) << (NQ - 1 - t);
    }

    for (int k = tid; k < DIM; k += 256) {
        float pr = 1.0f, pi = 0.0f;
        #pragma unroll
        for (int q = 0; q < NQ; ++q) {
            int rb = (r >> (NQ - 1 - q)) & 1;
            int cb = (k >> (NQ - 1 - q)) & 1;
            float gr = g[q][rb][cb][0], gi = g[q][rb][cb][1];
            float nr = pr * gr - pi * gi;
            pi = pr * gi + pi * gr;
            pr = nr;
        }
        Ur[n * DIM + k] = (_Float16)pr;
        Ui[n * DIM + k] = (_Float16)pi;
    }
}

// ---------------------------------------------------------------------------
// Kernel 2: fused  patch->psi (LDS)  ->  pipelined WMMA GEMM (real+imag)
//           ->  |.|^2  ->  Z contraction (LDS reduction)  ->  out[p][9]
// Workgroup = 32 patches x 512 columns; each wave32: 2 m-tiles x 4 n-tiles.
// ---------------------------------------------------------------------------
__global__ __launch_bounds__(256)
void qconv_kernel(const float* __restrict__ inp,
                  const _Float16* __restrict__ Ur,
                  const _Float16* __restrict__ Ui,
                  float* __restrict__ out)
{
    __shared__ float ampc[MROWS][NQ];
    __shared__ float amps[MROWS][NQ];
    // psi (f16, 33KB) and probs (f32, 66KB) live in the same region: psi is
    // dead once the GEMM loop finishes (barrier separates the phases).
    __shared__ __align__(16) char unionbuf[MROWS * PROB_LD * sizeof(float)];
    __shared__ float red[MROWS][8][NQ];

    _Float16 (*psiT)[PSI_LD]  = (_Float16 (*)[PSI_LD])unionbuf;
    float    (*probs)[PROB_LD] = (float (*)[PROB_LD])unionbuf;

    const int tid   = threadIdx.x;
    const int pbase = blockIdx.x * MROWS;

    // --- per-pixel half-angles: (cos(pi*x/2), sin(pi*x/2)) ------------------
    for (int idx = tid; idx < MROWS * NQ; idx += 256) {
        int m  = idx / NQ, q = idx % NQ;
        int p  = pbase + m;
        int b  = p >> 10, rem = p & 1023;
        int ho = rem >> 5, wo = rem & 31;
        int ky = q / 3,    kx = q % 3;
        float x  = inp[(b * 96 + ho * 3 + ky) * 96 + (wo * 3 + kx)];
        float ha = 1.5707963267948966f * x;
        ampc[m][q] = __cosf(ha);
        amps[m][q] = __sinf(ha);
    }
    __syncthreads();

    // --- product-state tile psi[32][512] (f16) ------------------------------
    {
        int m = tid >> 3;
        int jbase = (tid & 7) * 64;
        for (int jj = 0; jj < 64; ++jj) {
            int j = jbase + jj;
            float v = 1.0f;
            #pragma unroll
            for (int q = 0; q < NQ; ++q)
                v *= ((j >> (NQ - 1 - q)) & 1) ? amps[m][q] : ampc[m][q];
            psiT[m][j] = (_Float16)v;
        }
    }
    __syncthreads();

    // --- pipelined WMMA GEMM: C[m][n] = sum_k psi[m][k] * U[n][k] -----------
    const int wave  = tid >> 5;
    const int lane  = tid & 31;
    const int nr    = lane & 15;     // A: row-in-tile / B: col-in-tile
    const int khalf = lane >> 4;     // which K half this lane holds

    const v8f zero = {0.f, 0.f, 0.f, 0.f, 0.f, 0.f, 0.f, 0.f};
    v8f acc[2][4][2];                // [m-tile][n-tile][re/im]
    #pragma unroll
    for (int mt = 0; mt < 2; ++mt)
        #pragma unroll
        for (int nt = 0; nt < 4; ++nt) {
            acc[mt][nt][0] = zero;
            acc[mt][nt][1] = zero;
        }

    const _Float16* ubase[4][2];
    #pragma unroll
    for (int nt = 0; nt < 4; ++nt) {
        int nrow = (wave * 4 + nt) * 16 + nr;
        ubase[nt][0] = Ur + nrow * DIM;
        ubase[nt][1] = Ui + nrow * DIM;
    }

    auto loadA = [&](int mt, int ko) -> v16h {
        const _Float16* p = &psiT[mt * 16 + nr][ko];
        return make_frag(*(const v8h*)p, *(const v8h*)(p + 16));
    };
    auto loadB = [&](int nt, int pl, int ko) -> v16h {
        const _Float16* p = ubase[nt][pl] + ko;
        return make_frag(*(const v8h*)p, *(const v8h*)(p + 16));
    };

    // stage pipeline: prefetch next (n-tile, k-step) B pair while 4 WMMAs run
    v16h aC[2], aN[2], bCur[2], bNxt[2];
    {
        const int ko0 = khalf * 8;
        aC[0]   = loadA(0, ko0);
        aC[1]   = loadA(1, ko0);
        bCur[0] = loadB(0, 0, ko0);
        bCur[1] = loadB(0, 1, ko0);
    }

    for (int ks = 0; ks < DIM / 32; ++ks) {
        const int koc = ks * 32 + khalf * 8;
        const int kon = ((ks + 1) & (DIM / 32 - 1)) * 32 + khalf * 8;
        #pragma unroll
        for (int nt = 0; nt < 4; ++nt) {
            const int ntn = (nt + 1) & 3;
            const int kob = (nt < 3) ? koc : kon;
            bNxt[0] = loadB(ntn, 0, kob);
            bNxt[1] = loadB(ntn, 1, kob);
            if (nt == 3) {                       // A for next k-step
                aN[0] = loadA(0, kon);
                aN[1] = loadA(1, kon);
            }
            #pragma unroll
            for (int pl = 0; pl < 2; ++pl)
                #pragma unroll
                for (int mt = 0; mt < 2; ++mt)
                    acc[mt][nt][pl] = __builtin_amdgcn_wmma_f32_16x16x32_f16(
                        false, aC[mt], false, bCur[pl], (short)0,
                        acc[mt][nt][pl], false, false);
            bCur[0] = bNxt[0];
            bCur[1] = bNxt[1];
        }
        aC[0] = aN[0];
        aC[1] = aN[1];
    }

    // --- probabilities into the (reused) LDS region -------------------------
    __syncthreads();                 // all waves done reading psiT
    #pragma unroll
    for (int mt = 0; mt < 2; ++mt)
        #pragma unroll
        for (int nt = 0; nt < 4; ++nt) {
            const int ncol = (wave * 4 + nt) * 16 + nr;
            #pragma unroll
            for (int v = 0; v < 8; ++v) {
                int m = mt * 16 + v + 8 * khalf;   // C layout: vgpr->M, lane->N
                float cr = acc[mt][nt][0][v], ci = acc[mt][nt][1][v];
                probs[m][ncol] = cr * cr + ci * ci;
            }
        }
    __syncthreads();

    // --- Z contraction: expz[m][q] = sum_n probs[m][n]*(1 - 2*bit_q(n)) -----
    {
        int m = tid >> 3, col = tid & 7;
        float a2[NQ] = {0.f, 0.f, 0.f, 0.f, 0.f, 0.f, 0.f, 0.f, 0.f};
        for (int i = 0; i < 64; ++i) {
            int nn = i * 8 + col;
            float p = probs[m][nn];
            #pragma unroll
            for (int q = 0; q < NQ; ++q)
                a2[q] += ((nn >> (NQ - 1 - q)) & 1) ? -p : p;
        }
        #pragma unroll
        for (int q = 0; q < NQ; ++q) red[m][col][q] = a2[q];
    }
    __syncthreads();

    for (int idx = tid; idx < MROWS * NQ; idx += 256) {
        int m = idx / NQ, q = idx % NQ;
        float s = 0.f;
        #pragma unroll
        for (int c = 0; c < 8; ++c) s += red[m][c][q];
        out[(pbase + m) * NQ + q] = s;
    }
}

// ---------------------------------------------------------------------------
extern "C" void kernel_launch(void* const* d_in, const int* in_sizes, int n_in,
                              void* d_out, int out_size, void* d_ws, size_t ws_size,
                              hipStream_t stream)
{
    const float* inp    = (const float*)d_in[0];   // (16,96,96,1) f32
    const float* weight = (const float*)d_in[1];   // (27,) f32

    // workspace: Ur (512KB f16) | Ui (512KB f16)
    _Float16* Ur = (_Float16*)d_ws;
    _Float16* Ui = (_Float16*)((char*)d_ws + (size_t)DIM * DIM * sizeof(short));

    build_u_kernel<<<DIM, 256, 0, stream>>>(weight, Ur, Ui);

    // 16384 patches / 32 per workgroup = 512 workgroups
    qconv_kernel<<<512, 256, 0, stream>>>(inp, Ur, Ui, (float*)d_out);
}
